// CondInstDecoder_39350490366663
// MI455X (gfx1250) — matrix operand order repaired and checked by reading, not compiled, not run on hardware
//
#include <hip/hip_runtime.h>
#include <hip/hip_bf16.h>

typedef __attribute__((ext_vector_type(16))) _Float16 v16h;
typedef __attribute__((ext_vector_type(8)))  _Float16 v8h;
typedef __attribute__((ext_vector_type(8)))  float    v8f;
typedef __attribute__((ext_vector_type(4)))  float    v4f;

#define NPTS 5456
#define BATCH 4
#define TOPN 1000
#define MAXOBJ 100

// ---- workspace layout (bytes) ----
#define OFF_SCORES   0u          // float[B*N]
#define OFF_CLASSES  87296u      // int[B*N]
#define OFF_BOXES    174592u     // float[B*N*4]
#define OFF_TOPIDX   523776u     // int[B*1000]
#define OFF_TOPS     539776u     // float[B*1000]
#define OFF_FN       555776u     // int[B]

// ---- output layout (floats) ----
#define OUT_SCORES   0u
#define OUT_CLASSES  400u
#define OUT_MASKS    800u
#define OUT_BOXES    104858400u  // 800 + 4*100*512*512

// -------------------------------------------------------------------------
// Kernel 1: per-point score / class / box decode
// -------------------------------------------------------------------------
__global__ __launch_bounds__(256) void score_kernel(
    const float* __restrict__ cls, const float* __restrict__ reg,
    const float* __restrict__ center, const float* __restrict__ pos,
    float* __restrict__ ws_scores, int* __restrict__ ws_classes,
    float* __restrict__ ws_boxes, int total) {
  int t = blockIdx.x * 256 + threadIdx.x;
  if (t >= total) return;
  const float* cp = cls + (size_t)t * 80;
  float mx = cp[0]; int mi = 0;
  #pragma unroll 4
  for (int c = 1; c < 80; ++c) { float v = cp[c]; if (v > mx) { mx = v; mi = c; } }
  ws_scores[t] = sqrtf(mx * center[t]);
  ws_classes[t] = mi;
  float px = pos[2 * t], py = pos[2 * t + 1];
  const float* rp = reg + (size_t)t * 4;
  ws_boxes[4 * t + 0] = (float)(int)(px - rp[0]);
  ws_boxes[4 * t + 1] = (float)(int)(py - rp[1]);
  ws_boxes[4 * t + 2] = (float)(int)(px + rp[2]);
  ws_boxes[4 * t + 3] = (float)(int)(py + rp[3]);
}

// -------------------------------------------------------------------------
// Kernel 2: per-image full bitonic sort (8192 padded) in LDS -> top 1000
// key = monotonic(score)<<32 | ~idx   (descending sort => score desc, idx asc)
// -------------------------------------------------------------------------
__global__ __launch_bounds__(1024) void topk_kernel(
    const float* __restrict__ ws_scores,
    int* __restrict__ ws_topidx, float* __restrict__ ws_tops) {
  int b = blockIdx.x;
  int tid = threadIdx.x;
  __shared__ unsigned long long keys[8192];
  const float* sp = ws_scores + (size_t)b * NPTS;
  for (int i = tid; i < 8192; i += 1024) {
    unsigned long long kk = 0ull;
    if (i < NPTS) {
      float s = sp[i];
      s = (s > 0.1f) ? s : -1.0f;             // reference masks sub-threshold to -1
      unsigned u = __float_as_uint(s);
      u = (u & 0x80000000u) ? ~u : (u | 0x80000000u); // total-order transform
      kk = ((unsigned long long)u << 32) | (unsigned)(~(unsigned)i);
    }
    keys[i] = kk;
  }
  __syncthreads();
  for (int kseg = 2; kseg <= 8192; kseg <<= 1) {
    for (int j = kseg >> 1; j > 0; j >>= 1) {
      for (int base = tid; base < 4096; base += 1024) {
        int i = ((base & ~(j - 1)) << 1) | (base & (j - 1));
        int ixj = i | j;
        unsigned long long a = keys[i], c = keys[ixj];
        bool desc = ((i & kseg) == 0);
        if ((a < c) == desc) { keys[i] = c; keys[ixj] = a; }
      }
      __syncthreads();
    }
  }
  for (int q = tid; q < TOPN; q += 1024) {
    unsigned long long kk = keys[q];
    unsigned hi = (unsigned)(kk >> 32);
    float s = (hi & 0x80000000u) ? __uint_as_float(hi ^ 0x80000000u)
                                 : __uint_as_float(~hi);
    ws_topidx[b * TOPN + q] = (int)(~(unsigned)kk);
    ws_tops[b * TOPN + q] = s;
  }
}

// -------------------------------------------------------------------------
// Kernel 3: exact sequential greedy NMS keep-count (matches lax.scan)
// -------------------------------------------------------------------------
__global__ __launch_bounds__(1024) void nms_kernel(
    const float* __restrict__ ws_boxes, const int* __restrict__ ws_topidx,
    const float* __restrict__ ws_tops, int* __restrict__ ws_fn) {
  int b = blockIdx.x;
  int tid = threadIdx.x;
  __shared__ float bx0[TOPN], by0[TOPN], bx1[TOPN], by1[TOPN], area[TOPN];
  __shared__ unsigned char valid[TOPN], sup[TOPN];
  __shared__ int keepflag, kcount;
  for (int q = tid; q < TOPN; q += 1024) {
    int idx = ws_topidx[b * TOPN + q];
    const float* bp = ws_boxes + ((size_t)b * NPTS + idx) * 4;
    float x0 = bp[0], y0 = bp[1], x1 = bp[2], y1 = bp[3];
    bx0[q] = x0; by0[q] = y0; bx1[q] = x1; by1[q] = y1;
    float w = fmaxf(x1 - x0, 0.0f), h = fmaxf(y1 - y0, 0.0f);
    area[q] = w * h;
    valid[q] = (ws_tops[b * TOPN + q] > 0.1f) ? 1 : 0;
    sup[q] = 0;
  }
  if (tid == 0) kcount = 0;
  __syncthreads();
  for (int i = 0; i < TOPN; ++i) {
    if (tid == 0) {
      int kp = (valid[i] && !sup[i]) ? 1 : 0;
      kcount += kp; keepflag = kp;
    }
    __syncthreads();
    if (keepflag) {
      int j = i + 1 + tid;
      if (j < TOPN) {
        float lx = fmaxf(bx0[i], bx0[j]), ly = fmaxf(by0[i], by0[j]);
        float rx = fminf(bx1[i], bx1[j]), ry = fminf(by1[i], by1[j]);
        float w = fmaxf(rx - lx, 0.0f), h = fmaxf(ry - ly, 0.0f);
        float inter = w * h;
        float un = fmaxf(area[i] + area[j] - inter, 1e-9f);
        if (inter / un > 0.6f) sup[j] = 1;
      }
    }
    __syncthreads();
  }
  if (tid == 0) ws_fn[b] = (kcount > MAXOBJ) ? MAXOBJ : kcount;
}

// -------------------------------------------------------------------------
// Kernel 4: scores / classes / boxes outputs with slot masking
// -------------------------------------------------------------------------
__global__ __launch_bounds__(256) void finalize_kernel(
    const int* __restrict__ ws_topidx, const float* __restrict__ ws_tops,
    const int* __restrict__ ws_fn, const int* __restrict__ ws_classes,
    const float* __restrict__ ws_boxes, float* __restrict__ out) {
  int t = blockIdx.x * 256 + threadIdx.x;
  if (t >= BATCH * MAXOBJ) return;
  int b = t / MAXOBJ, m = t % MAXOBJ;
  bool slot = m < ws_fn[b];
  int idx = ws_topidx[b * TOPN + m];
  out[OUT_SCORES + t]  = slot ? ws_tops[b * TOPN + m] : -1.0f;
  out[OUT_CLASSES + t] = slot ? (float)ws_classes[b * NPTS + idx] : -1.0f;
  const float* bp = ws_boxes + ((size_t)b * NPTS + idx) * 4;
  float* ob = out + OUT_BOXES + (size_t)t * 4;
  ob[0] = slot ? bp[0] : 0.0f;
  ob[1] = slot ? bp[1] : 0.0f;
  ob[2] = slot ? bp[2] : 0.0f;
  ob[3] = slot ? bp[3] : 0.0f;
}

// -------------------------------------------------------------------------
// Kernel 5: per-object mask MLP via WMMA + nearest upsample + threshold.
// One block per (b,m) object, 256 threads = 8 waves, 32 tiles of 16 px/wave.
// Layer pipeline: A(16x32 f16) x B(32x16 f16) -> f32 via v_wmma_f32_16x16x32_f16
// -------------------------------------------------------------------------
__global__ __launch_bounds__(256) void mask_kernel(
    const float* __restrict__ ctrl_all, const float* __restrict__ mfeat,
    const int* __restrict__ ws_topidx, const int* __restrict__ ws_fn,
    float* __restrict__ out) {
  int ob = blockIdx.x;                 // 0..399
  int b = ob / MAXOBJ, m = ob % MAXOBJ;
  int tid = threadIdx.x;

  __shared__ float ldsc[176];                       // ctrl copy
  __shared__ __align__(16) _Float16 hstage[8][128]; // per-wave 16x8 f16 staging
  __shared__ unsigned char sgn[4096];               // 64x64 logit signs
  __shared__ unsigned char smap[512];               // upsample source map

  for (int j = tid; j < 512; j += 256) {
    int jm = j - 4; if (jm < 0) jm = 0;
    int s = (jm * 65) / 513; if (s > 63) s = 63;
    smap[j] = (unsigned char)s;
  }

  float* outp = out + OUT_MASKS + (size_t)ob * 262144;
  bool slot = m < ws_fn[b];
  if (!slot) {                                      // clear poisoned output
    v4f z = {0.f, 0.f, 0.f, 0.f};
    v4f* o4 = (v4f*)outp;
    for (int it = tid; it < 65536; it += 256)
      __builtin_nontemporal_store(z, &o4[it]);
    return;
  }

  int idx = ws_topidx[b * TOPN + m];
  const float* cg = ctrl_all + ((size_t)b * NPTS + idx) * 169;
  if (tid < 169) ldsc[tid] = cg[tid];
  __syncthreads();

  int lane = tid & 31;
  int wave = tid >> 5;
  int n16 = lane & 15;
  bool hiHalf = lane >= 16;

  // Build B matrices (w^T, zero-padded to 32x16) in this code's contiguous k order:
  // lane L: N = L%16, VGPR v holds K = (L<16?0:16)+2v, +1  => element e <-> K=base+e
  v16h B1 = {}; v16h B2 = {}; v16h B3 = {};
  if (!hiHalf && n16 < 8) {
    #pragma unroll
    for (int e = 0; e < 10; ++e) B1[e] = (_Float16)ldsc[n16 * 10 + e];      // w1[o][c]
    #pragma unroll
    for (int e = 0; e < 8; ++e)  B2[e] = (_Float16)ldsc[88 + n16 * 8 + e];  // w2[o][c]
  }
  if (lane == 0) {
    #pragma unroll
    for (int e = 0; e < 8; ++e)  B3[e] = (_Float16)ldsc[160 + e];           // w3[c]
  }
  float bias1 = (n16 < 8) ? ldsc[80 + n16] : 0.0f;
  float bias2 = (n16 < 8) ? ldsc[152 + n16] : 0.0f;
  float bias3 = ldsc[168];

  _Float16* hb = &hstage[wave][0];
  const float* mp = mfeat + (size_t)b * (64 * 64 * 8);

  for (int it = 0; it < 32; ++it) {
    int t = wave * 32 + it;                     // tile id, rows [16t, 16t+15]
    int row = t * 16 + n16;
    int y = row >> 6, x = row & 63;

    // A1 in contiguous k order (as B): lanes 0-15 hold K=0..7 (mask feats) in elems 0..7
    // and K=8..9 (rx, ry) in elems 8..9. Rest zero-padded.
    v16h A = {};
    if (!hiHalf) {
      const float* fp = mp + (size_t)row * 8;
      #pragma unroll
      for (int e = 0; e < 8; ++e) A[e] = (_Float16)fp[e];
      A[8] = (_Float16)((float)x * (2.0f / 63.0f) - 1.0f);
      A[9] = (_Float16)((float)y * (2.0f / 63.0f) - 1.0f);
    }

    v8f C = {};
    v8f D1 = __builtin_amdgcn_wmma_f32_16x16x32_f16(false, A, false, B1,
                                                    (short)0, C, false, false);
    // relu+bias, transpose D-layout -> A-layout via per-wave LDS staging
    if (n16 < 8) {
      int mb = hiHalf ? 8 : 0;
      #pragma unroll
      for (int i2 = 0; i2 < 8; ++i2) {
        float h = fmaxf(D1[i2] + bias1, 0.0f);
        hb[(mb + i2) * 8 + n16] = (_Float16)h;   // [m][c]
      }
    }
    __asm volatile("s_wait_dscnt 0" ::: "memory");
    v16h A2 = {};
    if (!hiHalf) {
      v8h hv = *(const v8h*)(hb + lane * 8);
      #pragma unroll
      for (int e = 0; e < 8; ++e) A2[e] = hv[e];
    }
    v8f D2 = __builtin_amdgcn_wmma_f32_16x16x32_f16(false, A2, false, B2,
                                                    (short)0, C, false, false);
    if (n16 < 8) {
      int mb = hiHalf ? 8 : 0;
      #pragma unroll
      for (int i2 = 0; i2 < 8; ++i2) {
        float h = fmaxf(D2[i2] + bias2, 0.0f);
        hb[(mb + i2) * 8 + n16] = (_Float16)h;
      }
    }
    __asm volatile("s_wait_dscnt 0" ::: "memory");
    v16h A3 = {};
    if (!hiHalf) {
      v8h hv = *(const v8h*)(hb + lane * 8);
      #pragma unroll
      for (int e = 0; e < 8; ++e) A3[e] = hv[e];
    }
    v8f D3 = __builtin_amdgcn_wmma_f32_16x16x32_f16(false, A3, false, B3,
                                                    (short)0, C, false, false);
    // logits live in column N=0: lanes 0 (M=i) and 16 (M=8+i)
    if (n16 == 0) {
      int mb = hiHalf ? 8 : 0;
      #pragma unroll
      for (int i2 = 0; i2 < 8; ++i2)
        sgn[t * 16 + mb + i2] = (D3[i2] + bias3 > 0.0f) ? 1 : 0;
    }
  }
  __syncthreads();

  // nearest upsample 64x64 -> 512x512, coalesced NT float4 stores
  v4f* o4 = (v4f*)outp;
  for (int it = 0; it < 256; ++it) {
    int p4 = (it * 256 + tid) * 4;
    int i = p4 >> 9, j = p4 & 511;
    int sy = (int)smap[i] << 6;
    v4f v;
    v.x = sgn[sy + smap[j + 0]] ? 1.0f : 0.0f;
    v.y = sgn[sy + smap[j + 1]] ? 1.0f : 0.0f;
    v.z = sgn[sy + smap[j + 2]] ? 1.0f : 0.0f;
    v.w = sgn[sy + smap[j + 3]] ? 1.0f : 0.0f;
    __builtin_nontemporal_store(v, &o4[p4 >> 2]);
  }
}

// -------------------------------------------------------------------------
extern "C" void kernel_launch(void* const* d_in, const int* in_sizes, int n_in,
                              void* d_out, int out_size, void* d_ws, size_t ws_size,
                              hipStream_t stream) {
  (void)in_sizes; (void)n_in; (void)out_size; (void)ws_size;
  const float* cls    = (const float*)d_in[0];  // (B,N,80)
  const float* reg    = (const float*)d_in[1];  // (B,N,4)
  const float* center = (const float*)d_in[2];  // (B,N,1)
  const float* ctrl   = (const float*)d_in[3];  // (B,N,169)
  const float* pos    = (const float*)d_in[4];  // (B,N,2)
  const float* mfeat  = (const float*)d_in[5];  // (B,64,64,8)
  float* out = (float*)d_out;
  char* ws = (char*)d_ws;
  float* ws_scores  = (float*)(ws + OFF_SCORES);
  int*   ws_classes = (int*)  (ws + OFF_CLASSES);
  float* ws_boxes   = (float*)(ws + OFF_BOXES);
  int*   ws_topidx  = (int*)  (ws + OFF_TOPIDX);
  float* ws_tops    = (float*)(ws + OFF_TOPS);
  int*   ws_fn      = (int*)  (ws + OFF_FN);

  int total = BATCH * NPTS;
  score_kernel<<<(total + 255) / 256, 256, 0, stream>>>(
      cls, reg, center, pos, ws_scores, ws_classes, ws_boxes, total);
  topk_kernel<<<BATCH, 1024, 0, stream>>>(ws_scores, ws_topidx, ws_tops);
  nms_kernel<<<BATCH, 1024, 0, stream>>>(ws_boxes, ws_topidx, ws_tops, ws_fn);
  finalize_kernel<<<(BATCH * MAXOBJ + 255) / 256, 256, 0, stream>>>(
      ws_topidx, ws_tops, ws_fn, ws_classes, ws_boxes, out);
  mask_kernel<<<BATCH * MAXOBJ, 256, 0, stream>>>(
      ctrl, mfeat, ws_topidx, ws_fn, out);
}
